// FilteredNoise_2310692406021
// MI455X (gfx1250) — matrix operand order repaired and checked by reading, not compiled, not run on hardware
//
#include <hip/hip_runtime.h>

// FilteredNoise on MI455X: DFT-as-GEMM driven by v_wmma_f32_16x16x32_f16.
//   Stage 0: generate forward/inverse DFT bases (f16), pre-swizzled to this code's contiguous-k
//            wave32 WMMA B-fragment order (one contiguous 32B load per lane).
//   Stage 1: Z = Frames(16016x1024) @ Ffwd(1024x1024)      [WMMA f16 -> f32 acc]
//            epilogue: Z' = Z * H[frame, band] * c_k  -> f16
//   Stage 2: Y = Z'(16016x1024) @ Binv(1024x1024)          [Hann window folded in]
//   Stage 3: overlap-add gather (4 taps/sample, no atomics -> deterministic)
// Register blocking: 32-frame M-block in LDS (64KB), 2 M-frags x 2 N-frags per
// wave -> 4 WMMAs per B-fragment fetch, halving basis re-read traffic from L2.

typedef __attribute__((ext_vector_type(16))) _Float16 v16h;
typedef __attribute__((ext_vector_type(8)))  _Float16 v8h;
typedef __attribute__((ext_vector_type(8)))  float    v8f;
typedef __attribute__((ext_vector_type(4)))  float    v4f;

#define NFFT     1024
#define HOP      256
#define NB       128
#define NBATCH   16
#define NT       1001
#define NFRAMES  (NBATCH * NT)   // 16016
#define LNOISE   256000
#define PAD      512
#define MBLK     32              // frames per workgroup (A tile rows)
#define MBLOCKS  501             // ceil(16016/32)

// basis value for spectral column j (0..1023) at time n:
//   k = (j&511)+1 ; j<512 -> cos(2*pi*k*n/1024), j>=512 -> -sin(2*pi*k*n/1024)
__device__ __forceinline__ float basis_val(int j, int n) {
    int k = (j & 511) + 1;
    int m = (k * n) & 1023;                       // exact modular angle reduction
    float a = (float)m * 6.135923151e-03f;        // 2*pi/1024
    return (j < 512) ? __cosf(a) : -__sinf(a);
}

// Write both bases directly in this code's wave32 B-fragment order (contiguous k = 16h + s):
//   halfs[ ((ctile*32 + ktile)*32 + lane)*16 + h ]
//   lane<16 : (K = ktile*32 + h,      N = ctile*16 + lane)
//   lane>=16: (K = ktile*32 + 16 + h, N = ctile*16 + lane - 16)
// => the storage address equals the linear element id r below.
__global__ void init_bases(_Float16* __restrict__ fwd_sw,
                           _Float16* __restrict__ binv_sw) {
    int id   = blockIdx.x * blockDim.x + threadIdx.x;   // 2 * 1M threads
    int sel  = id >> 20;
    int r    = id & 0xFFFFF;
    int tile = r >> 9;                                  // 0..2047
    int lane = (r >> 4) & 31;
    int h    = r & 15;
    int kt   = tile & 31;
    int ct   = tile >> 5;
    int kk   = kt * 32 + ((lane < 16) ? h : 16 + h);    // K index (0..1023)
    int cc   = ct * 16 + (lane & 15);                   // N index (0..1023)
    if (sel == 0) {
        // forward basis: K = time n, N = spectral j
        fwd_sw[r] = (_Float16)basis_val(cc, kk);
    } else {
        // inverse basis: K = spectral j, N = time n ; Hann window folded in
        float w = 0.5f * (1.0f - __cosf((float)cc * 6.141921126e-03f)); // 2*pi/1023
        binv_sw[r] = (_Float16)(basis_val(kk, cc) * w);
    }
}

// Load one 16x32 A fragment (contiguous k = 16h + s layout) from an LDS tile.
__device__ __forceinline__ v16h load_afrag(const _Float16* row_base, int aoff) {
    v8h alo = *(const v8h*)(row_base + aoff);            // K 0..7  / 16..23
    v8h ahi = *(const v8h*)(row_base + 8 + aoff);        // K 8..15 / 24..31
    return __builtin_shufflevector(alo, ahi,
               0,1,2,3,4,5,6,7,8,9,10,11,12,13,14,15);
}

// -------- Stage 1: forward DFT GEMM + band scaling --------------------------
// grid = (8, 501), block = 128 (4 waves). Wave wv owns column tiles
// jt0 = blockIdx.x*8 + wv*2 and jt0+1 (32 spectral columns), 2 M-frags, K=1024.
__global__ __launch_bounds__(128) void gemm_fwd(
    const float* __restrict__ noise, const float* __restrict__ fb,
    const _Float16* __restrict__ fwd_sw, _Float16* __restrict__ Zp)
{
    __shared__ _Float16 Atile[MBLK * 1024];             // 64 KB, shared by 4 waves
    const int tid   = threadIdx.x;
    const int mbase = blockIdx.y * MBLK;

    // Cooperative A-tile load: frames built on the fly from noise (scale+pad).
    for (int c = tid; c < MBLK * 128; c += 128) {        // chunks of 8 halfs
        int m  = c >> 7;
        int n0 = (c & 127) << 3;
        int f  = mbase + m;
        v8h o;
        if (f < NFRAMES) {
            int b = f / NT;
            int t = f - b * NT;
            int q = t * HOP + n0 - PAD;                  // noise index (mult of 8)
            const float* np = noise + (size_t)b * LNOISE;
            if (q >= 0 && q + 7 < LNOISE) {
                v4f x0 = *(const v4f*)(np + q);
                v4f x1 = *(const v4f*)(np + q + 4);
                #pragma unroll
                for (int i = 0; i < 4; ++i) {
                    o[i]     = (_Float16)(x0[i] * 2.0f - 1.0f);
                    o[i + 4] = (_Float16)(x1[i] * 2.0f - 1.0f);
                }
            } else {
                #pragma unroll
                for (int i = 0; i < 8; ++i) {
                    int s = q + i;
                    float v = (s >= 0 && s < LNOISE) ? np[s] * 2.0f - 1.0f : 0.0f;
                    o[i] = (_Float16)v;
                }
            }
        } else {
            o = (v8h)(_Float16)0.0f;                     // tail padding rows
        }
        *(v8h*)(&Atile[m * 1024 + n0]) = o;
    }
    __syncthreads();

    const int wv   = tid >> 5;
    const int lane = tid & 31;
    const int jt0  = blockIdx.x * 8 + wv * 2;
    const int mrow = lane & 15;
    const int aoff = (lane < 16) ? 0 : 16;               // contiguous K layout (k = 16h + s)
    const _Float16* bp0 = fwd_sw + (size_t)jt0 * 32 * 512 + lane * 16;
    const _Float16* bp1 = bp0 + (size_t)32 * 512;

    v8f acc00 = {}, acc01 = {}, acc10 = {}, acc11 = {};
    #pragma unroll 4
    for (int kt = 0; kt < 32; ++kt) {
        v16h b0 = *(const v16h*)(bp0 + (size_t)kt * 512);
        v16h b1 = *(const v16h*)(bp1 + (size_t)kt * 512);
        v16h a0 = load_afrag(&Atile[mrow * 1024 + kt * 32], aoff);
        v16h a1 = load_afrag(&Atile[(16 + mrow) * 1024 + kt * 32], aoff);
        acc00 = __builtin_amdgcn_wmma_f32_16x16x32_f16(false, a0, false, b0, (short)0, acc00, false, false);
        acc01 = __builtin_amdgcn_wmma_f32_16x16x32_f16(false, a0, false, b1, (short)0, acc01, false, false);
        acc10 = __builtin_amdgcn_wmma_f32_16x16x32_f16(false, a1, false, b0, (short)0, acc10, false, false);
        acc11 = __builtin_amdgcn_wmma_f32_16x16x32_f16(false, a1, false, b1, (short)0, acc11, false, false);
    }

    // Epilogue: Z'[f,j] = Z[f,j] * H[f, band(j)] * c_k  -> f16
    #pragma unroll
    for (int mf = 0; mf < 2; ++mf) {
        const v8f* accm = mf ? &acc10 : &acc00;          // [nf] pairs below
        #pragma unroll
        for (int g = 0; g < 8; ++g) {
            int m = mf * 16 + ((lane < 16) ? g : g + 8); // ISA 32-bit C/D layout
            int f = mbase + m;
            if (f >= NFRAMES) continue;
            int b = f / NT;
            int t = f - b * NT;
            const float* fbbt = fb + (size_t)b * NB * NT + t;
            #pragma unroll
            for (int nf = 0; nf < 2; ++nf) {
                int j    = (jt0 + nf) * 16 + (lane & 15);
                int jm   = j & 511;                      // k-1
                int band = jm >> 2;                      // (k-1)//4
                float ck = (jm == 511) ? (1.0f / 1024.0f) : (2.0f / 1024.0f);
                float H  = fbbt[(size_t)band * NT];
                float zv = (mf ? (nf ? acc11[g] : acc10[g])
                              : (nf ? acc01[g] : acc00[g]));
                Zp[(size_t)f * 1024 + j] = (_Float16)(zv * H * ck);
            }
        }
        (void)accm;
    }
}

// -------- Stage 2: inverse DFT GEMM (window folded into Binv) ---------------
__global__ __launch_bounds__(128) void gemm_inv(
    const _Float16* __restrict__ Zp, const _Float16* __restrict__ binv_sw,
    float* __restrict__ Y)
{
    __shared__ _Float16 Atile[MBLK * 1024];
    const int tid   = threadIdx.x;
    const int mbase = blockIdx.y * MBLK;

    for (int c = tid; c < MBLK * 128; c += 128) {
        int m  = c >> 7;
        int n0 = (c & 127) << 3;
        int f  = mbase + m;
        v8h o;
        if (f < NFRAMES) {
            o = *(const v8h*)(Zp + (size_t)f * 1024 + n0);
        } else {
            o = (v8h)(_Float16)0.0f;
        }
        *(v8h*)(&Atile[m * 1024 + n0]) = o;
    }
    __syncthreads();

    const int wv   = tid >> 5;
    const int lane = tid & 31;
    const int nt0  = blockIdx.x * 8 + wv * 2;
    const int mrow = lane & 15;
    const int aoff = (lane < 16) ? 0 : 16;
    const _Float16* bp0 = binv_sw + (size_t)nt0 * 32 * 512 + lane * 16;
    const _Float16* bp1 = bp0 + (size_t)32 * 512;

    v8f acc00 = {}, acc01 = {}, acc10 = {}, acc11 = {};
    #pragma unroll 4
    for (int kt = 0; kt < 32; ++kt) {
        v16h b0 = *(const v16h*)(bp0 + (size_t)kt * 512);
        v16h b1 = *(const v16h*)(bp1 + (size_t)kt * 512);
        v16h a0 = load_afrag(&Atile[mrow * 1024 + kt * 32], aoff);
        v16h a1 = load_afrag(&Atile[(16 + mrow) * 1024 + kt * 32], aoff);
        acc00 = __builtin_amdgcn_wmma_f32_16x16x32_f16(false, a0, false, b0, (short)0, acc00, false, false);
        acc01 = __builtin_amdgcn_wmma_f32_16x16x32_f16(false, a0, false, b1, (short)0, acc01, false, false);
        acc10 = __builtin_amdgcn_wmma_f32_16x16x32_f16(false, a1, false, b0, (short)0, acc10, false, false);
        acc11 = __builtin_amdgcn_wmma_f32_16x16x32_f16(false, a1, false, b1, (short)0, acc11, false, false);
    }

    #pragma unroll
    for (int mf = 0; mf < 2; ++mf) {
        #pragma unroll
        for (int g = 0; g < 8; ++g) {
            int m = mf * 16 + ((lane < 16) ? g : g + 8);
            int f = mbase + m;
            if (f >= NFRAMES) continue;
            #pragma unroll
            for (int nf = 0; nf < 2; ++nf) {
                int ncol = (nt0 + nf) * 16 + (lane & 15);
                float yv = (mf ? (nf ? acc11[g] : acc10[g])
                              : (nf ? acc01[g] : acc00[g]));
                Y[(size_t)f * 1024 + ncol] = yv;
            }
        }
    }
}

// -------- Stage 3: overlap-add gather (deterministic, no atomics) -----------
__global__ void overlap_add(const float* __restrict__ Y, float* __restrict__ out) {
    int o = blockIdx.x * 256 + threadIdx.x;
    if (o >= NBATCH * LNOISE) return;
    int b   = o / LNOISE;
    int p   = o - b * LNOISE;
    int pos = p + PAD;                                   // position in padded signal
    int th  = pos >> 8;                                  // floor(pos/HOP)
    float acc = 0.0f;
    #pragma unroll
    for (int dt = 0; dt < 4; ++dt) {
        int t = th - dt;
        if (t >= 0 && t < NT) {
            int n = pos - (t << 8);                      // 0..1023 guaranteed
            acc += Y[(size_t)(b * NT + t) * 1024 + n];
        }
    }
    out[o] = acc;
}

extern "C" void kernel_launch(void* const* d_in, const int* in_sizes, int n_in,
                              void* d_out, int out_size, void* d_ws, size_t ws_size,
                              hipStream_t stream) {
    (void)in_sizes; (void)n_in; (void)out_size;
    const float* fb    = (const float*)d_in[0];   // (16,1,128,1001) f32
    const float* noise = (const float*)d_in[1];   // (16,1,256000)   f32
    float* out = (float*)d_out;                   // (16,1,256000)   f32

    char* ws = (char*)d_ws;
    size_t off_fwd  = 0;
    size_t off_binv = off_fwd  + (size_t)1024 * 1024 * 2;        // 2 MB
    size_t off_zp   = off_binv + (size_t)1024 * 1024 * 2;        // 2 MB
    size_t off_y    = off_zp   + (size_t)NFRAMES * 1024 * 2;     // ~31 MB (f16)
    size_t need     = off_y    + (size_t)NFRAMES * 1024 * 4;     // ~63 MB (f32)
    if (ws_size < need) return;                                  // not enough scratch

    _Float16* fwd_sw  = (_Float16*)(ws + off_fwd);
    _Float16* binv_sw = (_Float16*)(ws + off_binv);
    _Float16* Zp      = (_Float16*)(ws + off_zp);
    float*    Y       = (float*)   (ws + off_y);

    init_bases<<<(2 * 1024 * 1024) / 256, 256, 0, stream>>>(fwd_sw, binv_sw);
    gemm_fwd  <<<dim3(8, MBLOCKS), 128, 0, stream>>>(noise, fb, fwd_sw, Zp);
    gemm_inv  <<<dim3(8, MBLOCKS), 128, 0, stream>>>(Zp, binv_sw, Y);
    overlap_add<<<(NBATCH * LNOISE + 255) / 256, 256, 0, stream>>>(Y, out);
}
